// RecursiveSimple_57234734187257
// MI455X (gfx1250) — matrix operand order repaired and checked by reading, not compiled, not run on hardware
//
#include <hip/hip_runtime.h>

typedef _Float16 half_t;
typedef __attribute__((ext_vector_type(16))) _Float16 v16h;
typedef __attribute__((ext_vector_type(8)))  _Float16 v8h;
typedef __attribute__((ext_vector_type(8)))  float    v8f;
typedef __attribute__((ext_vector_type(4)))  float    v4f;

#define NNODES   1023
#define NROWS_U  1024   // NNODES real rows + 1 dummy row (index 1023) for masked stores
#define FEATS    64
#define HID      64
#define PADU     72    // halves per node row: 144B stride -> 16B aligned, bank-conflict-free
#define PADW     200   // halves per WhT row: 400B stride -> 16B aligned, bank-conflict-free
#define NTHREADS 512
#define NWAVES   16

// A-tile fragment from global f32 row (16x32 f16 A layout):
// lane group g=lane>>4: halves 0..7 = K kb+8g .. +7, halves 8..15 = K kb+8g+16 .. +23
__device__ __forceinline__ v16h load_a_x(const float* __restrict__ xrow, int kb, int khalf) {
  const v4f* p = (const v4f*)(xrow + kb + khalf);
  v4f f0 = p[0], f1 = p[1];
  const v4f* q = (const v4f*)(xrow + kb + khalf + 16);
  v4f f2 = q[0], f3 = q[1];
  v16h a;
#pragma unroll
  for (int j = 0; j < 4; ++j) {
    a[j]      = (half_t)f0[j];
    a[4 + j]  = (half_t)f1[j];
    a[8 + j]  = (half_t)f2[j];
    a[12 + j] = (half_t)f3[j];
  }
  return a;
}

// Same A layout but sourced from an f16 row in LDS
__device__ __forceinline__ v16h load_a_lds(const half_t* row, int kb, int khalf) {
  const half_t* p = row + kb + khalf;
  v8h lo = *(const v8h*)p;
  v8h hi = *(const v8h*)(p + 16);
  return __builtin_shufflevector(lo, hi, 0,1,2,3,4,5,6,7,8,9,10,11,12,13,14,15);
}

// B-tile fragment (32x16 f16 B layout): lane n = lane&15,
// lane group g: halves 0..7 = K kb+8g .. +7, halves 8..15 = K kb+8g+16 .. +23, from transposed weights WT[n][k]
__device__ __forceinline__ v16h load_b_lds(const half_t* wT, int stride, int n, int kb, int koff16) {
  const half_t* p = wT + n * stride + kb + (koff16 >> 1);
  v8h lo = *(const v8h*)p;
  v8h hi = *(const v8h*)(p + 16);
  return __builtin_shufflevector(lo, hi, 0,1,2,3,4,5,6,7,8,9,10,11,12,13,14,15);
}

__global__ __launch_bounds__(NTHREADS)
void tree_rnn_fused_kernel(const float* __restrict__ x,
                           const float* __restrict__ Wu, const float* __restrict__ bu,
                           const float* __restrict__ Wh, const float* __restrict__ bh,
                           const float* __restrict__ Wp, const float* __restrict__ bp,
                           float* __restrict__ out) {
  extern __shared__ char smem_raw[];
  float*  s_bu = (float*)smem_raw;                 // 64 f32
  float*  s_bh = s_bu + 64;                        // 64 f32   (-> offset 512 total)
  half_t* s_wu = (half_t*)(smem_raw + 512);        // WuT [64][PADU]  (WuT[n][k] = Wu[k][n])
  half_t* s_wh = s_wu + 64 * PADU;                 // WhT [64][PADW]  (WhT[n][kg] = Wh[kg][n])
  half_t* s_u  = s_wh + 64 * PADW;                 // NROWS_U rows of PADU halves (u, then h)

  const int tid  = threadIdx.x;
  const int lane = tid & 31;
  const int wave = tid >> 5;
  const int b    = blockIdx.x;

  // ---- stage 0: convert weights to f16, transposed, into LDS ----
  for (int idx = tid; idx < 64 * 64; idx += NTHREADS) {
    int n = idx >> 6, k = idx & 63;
    s_wu[n * PADU + k] = (half_t)Wu[k * 64 + n];
  }
  for (int idx = tid; idx < 192 * 64; idx += NTHREADS) {
    int n = idx & 63, kg = idx >> 6;
    s_wh[n * PADW + kg] = (half_t)Wh[kg * 64 + n];
  }
  if (tid < 64)        s_bu[tid]      = bu[tid];
  else if (tid < 128)  s_bh[tid - 64] = bh[tid - 64];
  __syncthreads();

  const int mA     = lane & 15;       // A-row within tile loaded by this lane
  const int hgrp   = lane >> 4;       // half-wave group
  const int khalf  = hgrp * 8;        // A K sub-offset
  const int koff16 = hgrp * 16;       // B K sub-offset
  const int ncol   = lane & 15;       // output column within 16-wide N block

  // ---- stage 1: u = x @ Wu + bu for all 1023 nodes (leaves: h = u) ----
  // Hoist the 8 Wu^T fragments (constant across tiles) out of the tile loop.
  v16h Bu[2][4];                       // [k-step][n-block]
#pragma unroll
  for (int nb = 0; nb < 4; ++nb) {
    Bu[0][nb] = load_b_lds(s_wu, PADU, nb * 16 + ncol, 0,  koff16);
    Bu[1][nb] = load_b_lds(s_wu, PADU, nb * 16 + ncol, 32, koff16);
  }
  float biasU[4], biasH[4];
#pragma unroll
  for (int nb = 0; nb < 4; ++nb) { biasU[nb] = s_bu[nb * 16 + ncol]; biasH[nb] = s_bh[nb * 16 + ncol]; }

  const float* xb = x + (size_t)b * NNODES * FEATS;
  for (int t = wave; t < 64; t += NWAVES) {                 // 64 tiles of 16 rows
    int nodebase = t * 16;
    int nodeLd = nodebase + mA; if (nodeLd > NNODES - 1) nodeLd = NNODES - 1;
    const float* xrow = xb + (size_t)nodeLd * FEATS;
    v16h a0 = load_a_x(xrow, 0,  khalf);
    v16h a1 = load_a_x(xrow, 32, khalf);

    v8f c[4];
#pragma unroll
    for (int nb = 0; nb < 4; ++nb) {
#pragma unroll
      for (int r = 0; r < 8; ++r) c[nb][r] = biasU[nb];
    }
    // K-outer / N-inner: 4 back-to-back WMMAs on independent accumulators per k-step
#pragma unroll
    for (int nb = 0; nb < 4; ++nb)
      c[nb] = __builtin_amdgcn_wmma_f32_16x16x32_f16(false, a0, false, Bu[0][nb], (short)0, c[nb], false, false);
#pragma unroll
    for (int nb = 0; nb < 4; ++nb)
      c[nb] = __builtin_amdgcn_wmma_f32_16x16x32_f16(false, a1, false, Bu[1][nb], (short)0, c[nb], false, false);

    // Unconditional stores: the only out-of-range row index is exactly 1023,
    // which is the dummy row of s_u.
#pragma unroll
    for (int nb = 0; nb < 4; ++nb) {
#pragma unroll
      for (int r = 0; r < 8; ++r) {
        int node = nodebase + r + 8 * hgrp;                 // C/D layout row (<= 1023)
        s_u[node * PADU + nb * 16 + ncol] = (half_t)c[nb][r];
      }
    }
  }
  __syncthreads();

  // ---- stage 2: bottom-up level sweep, h_i = relu([u_i,h_L,h_R] @ Wh + bh), in-place in LDS ----
  for (int d = 8; d >= 0; --d) {
    int nrows   = 1 << d;
    int lvlbase = nrows - 1;
    int ntiles  = (nrows + 15) >> 4;
    for (int t = wave; t < ntiles; t += NWAVES) {
      int rowbase = t * 16;
      int il = rowbase + mA; if (il > nrows - 1) il = nrows - 1;  // clamp loads (EXEC stays full)
      int node = lvlbase + il;
      const half_t* seg0 = s_u + (size_t)node * PADU;             // u_i
      const half_t* seg1 = s_u + (size_t)(2 * node + 1) * PADU;   // h_left
      const half_t* seg2 = s_u + (size_t)(2 * node + 2) * PADU;   // h_right
      v16h A[6];
      A[0] = load_a_lds(seg0, 0, khalf);  A[1] = load_a_lds(seg0, 32, khalf);
      A[2] = load_a_lds(seg1, 0, khalf);  A[3] = load_a_lds(seg1, 32, khalf);
      A[4] = load_a_lds(seg2, 0, khalf);  A[5] = load_a_lds(seg2, 32, khalf);

      v8f c[4];
#pragma unroll
      for (int nb = 0; nb < 4; ++nb) {
#pragma unroll
        for (int r = 0; r < 8; ++r) c[nb][r] = biasH[nb];
      }
      // K-outer / N-inner: per k-step, 4 independent-accumulator WMMAs back-to-back
#pragma unroll
      for (int ks = 0; ks < 6; ++ks) {
        v16h bm[4];
#pragma unroll
        for (int nb = 0; nb < 4; ++nb)
          bm[nb] = load_b_lds(s_wh, PADW, nb * 16 + ncol, ks * 32, koff16);
#pragma unroll
        for (int nb = 0; nb < 4; ++nb)
          c[nb] = __builtin_amdgcn_wmma_f32_16x16x32_f16(false, A[ks], false, bm[nb], (short)0, c[nb], false, false);
      }

      // Branchless masked store tail: invalid rows go to the dummy row (index NNODES).
#pragma unroll
      for (int nb = 0; nb < 4; ++nb) {
#pragma unroll
        for (int r = 0; r < 8; ++r) {
          int rr = rowbase + r + 8 * hgrp;
          int nodeSt = (rr < nrows) ? (lvlbase + rr) : NNODES;    // cndmask, no exec games
          float v = c[nb][r] > 0.0f ? c[nb][r] : 0.0f;            // relu
          s_u[nodeSt * PADU + nb * 16 + ncol] = (half_t)v;
        }
      }
    }
    __syncthreads();   // next level reads this level's h
  }

  // ---- stage 3: logits = h_root @ Wp + bp ----
  if (tid == 0) {
    float acc = bp[0];
#pragma unroll
    for (int k = 0; k < 64; ++k) acc += (float)s_u[k] * Wp[k];
    out[b] = acc;
  }
}

extern "C" void kernel_launch(void* const* d_in, const int* in_sizes, int n_in,
                              void* d_out, int out_size, void* d_ws, size_t ws_size,
                              hipStream_t stream) {
  (void)in_sizes; (void)n_in; (void)d_ws; (void)ws_size; (void)out_size;
  const float* x  = (const float*)d_in[0];
  const float* Wu = (const float*)d_in[1];
  const float* bu = (const float*)d_in[2];
  const float* Wh = (const float*)d_in[3];
  const float* bh = (const float*)d_in[4];
  const float* Wp = (const float*)d_in[5];
  const float* bp = (const float*)d_in[6];
  float* out = (float*)d_out;

  size_t shmem = 512                          // biases (f32)
               + (size_t)64 * PADU * 2        // WuT
               + (size_t)64 * PADW * 2        // WhT
               + (size_t)NROWS_U * PADU * 2;  // tree u/h buffer + dummy row
  // = 182,784 bytes < 320 KB per workgroup (CDNA5 WGP LDS)

  tree_rnn_fused_kernel<<<1024, NTHREADS, shmem, stream>>>(x, Wu, bu, Wh, bh, Wp, bp, out);
}
